// Model_3934190044152
// MI455X (gfx1250) — matrix operand order repaired and checked by reading, not compiled, not run on hardware
//
#include <hip/hip_runtime.h>
#include <hip/hip_bf16.h>

// ---------------------------------------------------------------------------
// CDNA5 (gfx1250) criss-cross attention, bf16 WMMA path.
// ---------------------------------------------------------------------------

typedef __attribute__((ext_vector_type(16))) __bf16 v16bf;
typedef __attribute__((ext_vector_type(8)))  __bf16 v8bf;
typedef __attribute__((ext_vector_type(8)))  float  v8f;

#define BATCH 8
#define CCH   256
#define CQK   32
#define HH    128
#define WW    128

// ---- WMMA helpers ----------------------------------------------------------

__device__ __forceinline__ v8f wmma_bf16(v16bf a, v16bf b, v8f c) {
  // v_wmma_f32_16x16x32_bf16  D = A(16x32) * B(32x16) + C
  return __builtin_amdgcn_wmma_f32_16x16x32_bf16(
      /*neg_a=*/false, a, /*neg_b=*/false, b,
      /*c_mod=*/(short)0, c, /*reuse_a=*/false, /*reuse_b=*/false);
}

// A fragment: 16x32 bf16, row-major with row stride `ld` (elements).
// Lane l: row m = row0 + (l&15); holds K runs [8*kh, 8*kh+8) and [16+8*kh, ...)
__device__ __forceinline__ v16bf load_frag_a(const __bf16* p, int row0, int ld, int k0) {
  int l = threadIdx.x & 31;
  const __bf16* q = p + (size_t)(row0 + (l & 15)) * (size_t)ld + k0 + ((l >> 4) << 3);
  v8bf lo = *(const v8bf*)(q);
  v8bf hi = *(const v8bf*)(q + 16);
  return __builtin_shufflevector(lo, hi, 0,1,2,3,4,5,6,7,8,9,10,11,12,13,14,15);
}

// B fragment: 32x16 bf16, but stored transposed (N x K row-major, stride `ld`).
// Lane l: col n = col0 + (l&15); holds K runs [8*kh, 8*kh+8) and [16+8*kh, ...)
__device__ __forceinline__ v16bf load_frag_b(const __bf16* p, int col0, int ld, int k0) {
  int l = threadIdx.x & 31;
  const __bf16* q = p + (size_t)(col0 + (l & 15)) * (size_t)ld + k0 + ((l >> 4) << 3);
  v8bf lo = *(const v8bf*)(q);
  v8bf hi = *(const v8bf*)(q + 16);
  return __builtin_shufflevector(lo, hi, 0,1,2,3,4,5,6,7,8,9,10,11,12,13,14,15);
}

// ---- K0: pack weights to bf16 (320x256 row-major: q rows 0-31, k 32-63, v 64-319)
__global__ void k0_prep(const float* __restrict__ Wq, const float* __restrict__ Wk,
                        const float* __restrict__ Wv, const float* __restrict__ bq,
                        const float* __restrict__ bk, const float* __restrict__ bv,
                        __bf16* __restrict__ Wall, float* __restrict__ ball) {
  int idx = blockIdx.x * 256 + threadIdx.x;
  if (idx < 320 * 256) {
    int m = idx >> 8, c = idx & 255;
    float w = (m < 32) ? Wq[m * 256 + c]
            : (m < 64) ? Wk[(m - 32) * 256 + c]
                       : Wv[(size_t)(m - 64) * 256 + c];
    Wall[idx] = (__bf16)w;
  }
  if (idx < 320) {
    ball[idx] = (idx < 32) ? bq[idx] : (idx < 64) ? bk[idx - 32] : bv[idx - 64];
  }
}

// ---- K1: QKV projection. One workgroup per (b, y): GEMM M=320, N=128(x), K=256.
// Outputs: q,k as (b, y, x, c[32]) bf16 ; v as (b, y, c[256], x) bf16.
__global__ void k1_qkv(const float* __restrict__ x, const __bf16* __restrict__ Wall,
                       const float* __restrict__ ball, __bf16* __restrict__ qb,
                       __bf16* __restrict__ kb, __bf16* __restrict__ vb) {
  extern __shared__ char smem[];
  __bf16* xt = (__bf16*)smem;        // [128][264]  (N x K, padded)
  const int LDX = 264;
  int b = blockIdx.x >> 7;
  int y = blockIdx.x & 127;
  int t = threadIdx.x;

  const float* xp = x + (size_t)b * CCH * (HH * WW) + (size_t)y * WW;
  for (int idx = t; idx < CCH * WW; idx += 256) {
    int c = idx >> 7, xw = idx & 127;
    xt[xw * LDX + c] = (__bf16)xp[(size_t)c * (HH * WW) + xw];
  }
  __syncthreads();

  int l = t & 31, wv = t >> 5;
  int kh = l >> 4, ln = l & 15;
  int n0 = wv * 16;                  // x-tile owned by this wave
  int n  = n0 + ln;
  for (int mt = 0; mt < 20; ++mt) {
    v8f acc = {};
#pragma unroll
    for (int ks = 0; ks < 8; ++ks) {
      v16bf a = load_frag_a(Wall, mt * 16, 256, ks * 32);
      v16bf bm = load_frag_b(xt, n0, LDX, ks * 32);
      acc = wmma_bf16(a, bm, acc);
    }
#pragma unroll
    for (int r = 0; r < 8; ++r) {
      int m = mt * 16 + r + kh * 8;
      float val = acc[r] + ball[m];
      if (m < 64) {
        __bf16* dst = (m < 32) ? qb : kb;
        int c = m & 31;
        dst[(((size_t)b * HH + y) * WW + n) * CQK + c] = (__bf16)val;
      } else {
        int c = m - 64;
        vb[(((size_t)b * HH + y) * CCH + c) * WW + n] = (__bf16)val;
      }
    }
  }
}

// ---- K2: v (b,y,c,x) -> vcol (b,x,c,y) via LDS tile transpose.
__global__ void k2_vcol(const __bf16* __restrict__ vrow, __bf16* __restrict__ vcol) {
  __shared__ __bf16 tile[32][33];
  int bid = blockIdx.x;
  int xt = bid & 3;  bid >>= 2;
  int yt = bid & 3;  bid >>= 2;
  int c  = bid & 255; bid >>= 8;
  int b  = bid;                      // grid = 8*256*4*4 = 32768
  int t = threadIdx.x, tx = t & 31, ty = t >> 5;
  size_t base = (size_t)b * (HH * CCH * WW) + (size_t)c * WW;
#pragma unroll
  for (int i = 0; i < 4; ++i) {
    int yy = yt * 32 + ty + i * 8;
    tile[ty + i * 8][tx] = vrow[base + (size_t)yy * (CCH * WW) + xt * 32 + tx];
  }
  __syncthreads();
#pragma unroll
  for (int i = 0; i < 4; ++i) {
    int xx = xt * 32 + ty + i * 8;
    vcol[base + (size_t)xx * (CCH * HH) + yt * 32 + tx] = tile[tx][ty + i * 8];
  }
}

// ---- K3/K4: attention logits + per-row softmax stats (flash-style, unnormalized).
// mode 0 (H pass): per (b,x) column, rows strided by 128*32; diag masked.
// mode 1 (W pass): per (b,y) row, rows strided by 32 (contiguous block).
// Writes P = exp(S - rowmax) in bf16 at att[(b,p,i,j)], stats mv/sv at (b,p,i).
__global__ void k34_logits(const __bf16* __restrict__ qb, const __bf16* __restrict__ kb,
                           __bf16* __restrict__ att, float* __restrict__ mv,
                           float* __restrict__ sv, int mode) {
  extern __shared__ char smem[];
  float* S = (float*)smem;           // [128][129]
  int b = blockIdx.x >> 7;
  int p = blockIdx.x & 127;
  size_t base;
  int rstride;
  if (mode == 0) { base = (size_t)b * (HH * WW * CQK) + (size_t)p * CQK; rstride = WW * CQK; }
  else           { base = (size_t)b * (HH * WW * CQK) + (size_t)p * (WW * CQK); rstride = CQK; }
  const __bf16* Q = qb + base;
  const __bf16* K = kb + base;

  int t = threadIdx.x, l = t & 31, wv = t >> 5;
  int kh = l >> 4, ln = l & 15;
  v16bf bf = load_frag_b(K, wv * 16, rstride, 0);   // j-tile owned by this wave
#pragma unroll
  for (int mt = 0; mt < 8; ++mt) {
    v16bf af = load_frag_a(Q, mt * 16, rstride, 0);
    v8f acc = {};
    acc = wmma_bf16(af, bf, acc);
    int j = wv * 16 + ln;
#pragma unroll
    for (int r = 0; r < 8; ++r) {
      int i = mt * 16 + r + kh * 8;
      S[i * 129 + j] = acc[r];
    }
  }
  __syncthreads();

  if (t < 128) {
    int i = t;
    float m = -3.0e38f;
    for (int j = 0; j < 128; ++j) {
      float v = S[i * 129 + j];
      if (mode == 0 && j == i) v = -1.0e30f;       // diag mask on H logits
      S[i * 129 + j] = v;
      m = fmaxf(m, v);
    }
    float s = 0.0f;
    __bf16* arow = att + (((size_t)b * 128 + p) * 128 + i) * 128;
    for (int j = 0; j < 128; ++j) {
      float e = __expf(S[i * 129 + j] - m);
      s += e;
      arow[j] = (__bf16)e;
    }
    size_t si = ((size_t)b * 128 + p) * 128 + i;
    mv[si] = m;
    sv[si] = s;
  }
}

// ---- K4b: combine the two softmax halves into per-pixel scales (gamma folded in).
__global__ void k4b_alpha(const float* __restrict__ mH, const float* __restrict__ sH,
                          const float* __restrict__ mW, const float* __restrict__ sW,
                          const float* __restrict__ gamma, float* __restrict__ aH,
                          float* __restrict__ aW) {
  int idx = blockIdx.x * 256 + threadIdx.x;
  if (idx >= BATCH * HH * WW) return;
  int b = idx >> 14, rem = idx & 16383, y = rem >> 7, x = rem & 127;
  size_t ih = ((size_t)b * 128 + x) * 128 + y;   // (b,x,y)
  size_t iw = ((size_t)b * 128 + y) * 128 + x;   // (b,y,x)
  float mh = mH[ih], mw = mW[iw];
  float m = fmaxf(mh, mw);
  float eh = __expf(mh - m), ew = __expf(mw - m);
  float Z = sH[ih] * eh + sW[iw] * ew;
  float g = gamma[0] / Z;
  aH[ih] = g * eh;
  aW[iw] = g * ew;
}

// ---- K5: outH. Per (b,x): Out[c,y] = sum_j Vcol[c,j] * P_H[y,j]; d_out = aH*Out.
__global__ void k5_outH(const __bf16* __restrict__ vcol, const __bf16* __restrict__ attH,
                        const float* __restrict__ aH, float* __restrict__ out) {
  int b = blockIdx.x >> 7, x = blockIdx.x & 127;
  int t = threadIdx.x, l = t & 31, wv = t >> 5;
  int kh = l >> 4, ln = l & 15;
  const __bf16* V = vcol + ((size_t)b * 128 + x) * (CCH * HH);  // (c, j=y) ld=128
  const __bf16* A = attH + ((size_t)b * 128 + x) * (HH * HH);   // (i=y, j) ld=128
  int n0 = wv * 16;
  v16bf b0 = load_frag_b(A, n0, 128, 0);
  v16bf b1 = load_frag_b(A, n0, 128, 32);
  v16bf b2 = load_frag_b(A, n0, 128, 64);
  v16bf b3 = load_frag_b(A, n0, 128, 96);
  int y = n0 + ln;
  float scale = aH[((size_t)b * 128 + x) * 128 + y];
  for (int mt = 0; mt < 16; ++mt) {
    v8f acc = {};
    acc = wmma_bf16(load_frag_a(V, mt * 16, 128, 0),  b0, acc);
    acc = wmma_bf16(load_frag_a(V, mt * 16, 128, 32), b1, acc);
    acc = wmma_bf16(load_frag_a(V, mt * 16, 128, 64), b2, acc);
    acc = wmma_bf16(load_frag_a(V, mt * 16, 128, 96), b3, acc);
#pragma unroll
    for (int r = 0; r < 8; ++r) {
      int c = mt * 16 + r + kh * 8;
      out[(((size_t)b * CCH + c) * HH + y) * WW + x] = scale * acc[r];
    }
  }
}

// ---- K6: outW + residual. Per (b,y): Out[c,x] = sum_j Vrow[c,j] * P_W[x,j];
//          d_out += aW*Out + x_in  (coalesced read-modify-write).
__global__ void k6_outW(const __bf16* __restrict__ vrow, const __bf16* __restrict__ attW,
                        const float* __restrict__ aW, const float* __restrict__ xin,
                        float* __restrict__ out) {
  int b = blockIdx.x >> 7, y = blockIdx.x & 127;
  int t = threadIdx.x, l = t & 31, wv = t >> 5;
  int kh = l >> 4, ln = l & 15;
  const __bf16* V = vrow + ((size_t)b * 128 + y) * (CCH * WW);  // (c, j=x) ld=128
  const __bf16* A = attW + ((size_t)b * 128 + y) * (WW * WW);   // (i=x, j) ld=128
  int n0 = wv * 16;
  v16bf b0 = load_frag_b(A, n0, 128, 0);
  v16bf b1 = load_frag_b(A, n0, 128, 32);
  v16bf b2 = load_frag_b(A, n0, 128, 64);
  v16bf b3 = load_frag_b(A, n0, 128, 96);
  int xw = n0 + ln;
  float scale = aW[((size_t)b * 128 + y) * 128 + xw];
  for (int mt = 0; mt < 16; ++mt) {
    __builtin_prefetch(V + (size_t)(mt * 16 + 16) * 128, 0, 1);  // global_prefetch
    v8f acc = {};
    acc = wmma_bf16(load_frag_a(V, mt * 16, 128, 0),  b0, acc);
    acc = wmma_bf16(load_frag_a(V, mt * 16, 128, 32), b1, acc);
    acc = wmma_bf16(load_frag_a(V, mt * 16, 128, 64), b2, acc);
    acc = wmma_bf16(load_frag_a(V, mt * 16, 128, 96), b3, acc);
#pragma unroll
    for (int r = 0; r < 8; ++r) {
      int c = mt * 16 + r + kh * 8;
      size_t o = (((size_t)b * CCH + c) * HH + y) * WW + xw;
      out[o] = out[o] + scale * acc[r] + xin[o];
    }
  }
}

// ---------------------------------------------------------------------------

extern "C" void kernel_launch(void* const* d_in, const int* in_sizes, int n_in,
                              void* d_out, int out_size, void* d_ws, size_t ws_size,
                              hipStream_t stream) {
  (void)in_sizes; (void)n_in; (void)out_size; (void)ws_size;
  const float* x     = (const float*)d_in[0];
  const float* Wq    = (const float*)d_in[1];
  const float* bq    = (const float*)d_in[2];
  const float* Wk    = (const float*)d_in[3];
  const float* bk    = (const float*)d_in[4];
  const float* Wv    = (const float*)d_in[5];
  const float* bv    = (const float*)d_in[6];
  const float* gamma = (const float*)d_in[7];
  float* out = (float*)d_out;

  // Workspace carve-out (256B aligned): ~220 MB total.
  char* p = (char*)d_ws;
  auto alloc = [&](size_t bytes) -> char* {
    char* r = p;
    p += (bytes + 255) & ~(size_t)255;
    return r;
  };
  const size_t NPIX = (size_t)BATCH * HH * WW;                   // 131072
  __bf16* Wall = (__bf16*)alloc(320 * 256 * sizeof(__bf16));
  float*  ball = (float*) alloc(320 * sizeof(float));
  __bf16* qb   = (__bf16*)alloc(NPIX * CQK * sizeof(__bf16));    // (b,y,x,c)
  __bf16* kb   = (__bf16*)alloc(NPIX * CQK * sizeof(__bf16));
  __bf16* vrow = (__bf16*)alloc(NPIX * CCH * sizeof(__bf16));    // (b,y,c,x)
  __bf16* vcol = (__bf16*)alloc(NPIX * CCH * sizeof(__bf16));    // (b,x,c,y)
  __bf16* attH = (__bf16*)alloc(NPIX * HH * sizeof(__bf16));     // (b,x,i,j)
  __bf16* attW = (__bf16*)alloc(NPIX * WW * sizeof(__bf16));     // (b,y,i,j)
  float*  mH   = (float*) alloc(NPIX * sizeof(float));
  float*  sH   = (float*) alloc(NPIX * sizeof(float));
  float*  mW   = (float*) alloc(NPIX * sizeof(float));
  float*  sW   = (float*) alloc(NPIX * sizeof(float));
  float*  aH   = (float*) alloc(NPIX * sizeof(float));
  float*  aW   = (float*) alloc(NPIX * sizeof(float));

  const int NBY = BATCH * HH;   // 1024 workgroups for per-row / per-col kernels

  k0_prep<<<320, 256, 0, stream>>>(Wq, Wk, Wv, bq, bk, bv, Wall, ball);
  k1_qkv<<<NBY, 256, 128 * 264 * sizeof(__bf16), stream>>>(x, Wall, ball, qb, kb, vrow);
  k2_vcol<<<BATCH * CCH * 16, 256, 0, stream>>>(vrow, vcol);
  k34_logits<<<NBY, 256, 128 * 129 * sizeof(float), stream>>>(qb, kb, attH, mH, sH, 0);
  k34_logits<<<NBY, 256, 128 * 129 * sizeof(float), stream>>>(qb, kb, attW, mW, sW, 1);
  k4b_alpha<<<(int)(NPIX / 256), 256, 0, stream>>>(mH, sH, mW, sW, gamma, aH, aW);
  k5_outH<<<NBY, 256, 0, stream>>>(vcol, attH, aH, out);
  k6_outW<<<NBY, 256, 0, stream>>>(vrow, attW, aW, x, out);
}
